// AlignmentUniformityLoss_65850438582800
// MI455X (gfx1250) — matrix operand order repaired and checked by reading, not compiled, not run on hardware
//
#include <hip/hip_runtime.h>

typedef __attribute__((ext_vector_type(16))) __bf16 v16bf;
typedef __attribute__((ext_vector_type(8)))  float  v8f;

#define N_ROWS 8192
#define K_DIM  512
#define KSTEPS (K_DIM / 32)
#define FRAG_CNT (N_ROWS * K_DIM / 8)   // uint4 (8 bf16) granules per plane

union FragU { unsigned u[8]; v16bf v; };

// Convert 8 consecutive f32 -> packed bf16 hi (truncate) + bf16 lo (residual).
__device__ __forceinline__ void cvt8(const float* __restrict__ p,
                                     unsigned* __restrict__ uh,
                                     unsigned* __restrict__ ul) {
  const float4 a = *reinterpret_cast<const float4*>(p);
  const float4 b = *reinterpret_cast<const float4*>(p + 4);
  float f[8] = {a.x, a.y, a.z, a.w, b.x, b.y, b.z, b.w};
#pragma unroll
  for (int e = 0; e < 4; ++e) {
    unsigned b0 = __float_as_uint(f[2 * e]);
    unsigned b1 = __float_as_uint(f[2 * e + 1]);
    unsigned h0 = b0 & 0xFFFF0000u;
    unsigned h1 = b1 & 0xFFFF0000u;
    float r0 = f[2 * e]     - __uint_as_float(h0);
    float r1 = f[2 * e + 1] - __uint_as_float(h1);
    uh[e] = (h0 >> 16) | h1;
    ul[e] = (__float_as_uint(r0) >> 16) | (__float_as_uint(r1) & 0xFFFF0000u);
  }
}

__device__ __forceinline__ v16bf frag_from(const uint4 a, const uint4 b) {
  FragU f;
  f.u[0] = a.x; f.u[1] = a.y; f.u[2] = a.z; f.u[3] = a.w;
  f.u[4] = b.x; f.u[5] = b.y; f.u[6] = b.z; f.u[7] = b.w;
  return f.v;
}

__global__ __launch_bounds__(32) void init_kernel(float* __restrict__ accum) {
  if (threadIdx.x < 4) accum[threadIdx.x] = 0.0f;
}

// One-time f32 -> (bf16 hi, bf16 lo) plane split; 8 elements per thread.
__global__ __launch_bounds__(256) void split_kernel(const float* __restrict__ src,
                                                    uint4* __restrict__ hi,
                                                    uint4* __restrict__ lo) {
  const int idx = blockIdx.x * 256 + threadIdx.x;
  if (idx >= FRAG_CNT) return;
  unsigned uh[4], ul[4];
  cvt8(src + (size_t)idx * 8, uh, ul);
  hi[idx] = make_uint4(uh[0], uh[1], uh[2], uh[3]);
  lo[idx] = make_uint4(ul[0], ul[1], ul[2], ul[3]);
}

// One block per row: row squared norms for x and y, plus align-loss partial sum.
__global__ __launch_bounds__(128) void norm_align_kernel(
    const float* __restrict__ X, const float* __restrict__ Y,
    float* __restrict__ nX, float* __restrict__ nY, float* __restrict__ accum) {
  const int row = blockIdx.x;
  const float* px = X + (size_t)row * K_DIM;
  const float* py = Y + (size_t)row * K_DIM;
  float sx = 0.0f, sy = 0.0f, sd = 0.0f;
  for (int c = threadIdx.x; c < K_DIM; c += 128) {
    float xv = px[c], yv = py[c];
    sx += xv * xv;
    sy += yv * yv;
    float d = xv - yv;
    sd += d * d;
  }
#pragma unroll
  for (int off = 16; off > 0; off >>= 1) {
    sx += __shfl_down(sx, off, 32);
    sy += __shfl_down(sy, off, 32);
    sd += __shfl_down(sd, off, 32);
  }
  __shared__ float red[3][4];
  const int wave = threadIdx.x >> 5, lane = threadIdx.x & 31;
  if (lane == 0) { red[0][wave] = sx; red[1][wave] = sy; red[2][wave] = sd; }
  __syncthreads();
  if (threadIdx.x == 0) {
    sx = red[0][0] + red[0][1] + red[0][2] + red[0][3];
    sy = red[1][0] + red[1][1] + red[1][2] + red[1][3];
    sd = red[2][0] + red[2][1] + red[2][2] + red[2][3];
    nX[row] = sx;
    nY[row] = sy;
    atomicAdd(accum + 2, sd);
  }
}

// 128x128 Gram tile per workgroup. LDS-staged operands (Ah|Al|Bh|Bl, 32KB per
// k-step, loaded once per block -> 3x less L2 traffic, coalesced 64B segments),
// register prefetch of the next k-step overlapping compute. Split-precision
// WMMA (Ah*Bh + Ah*Bl + Al*Bh) fused with exp(-2*d2) sum reduction.
// Symmetry: only jt >= it tiles; off-diagonal tiles weighted 2x.
__global__ __launch_bounds__(256) void gram_exp_kernel(
    const uint4* __restrict__ Xh, const uint4* __restrict__ Xl,
    const uint4* __restrict__ Yh, const uint4* __restrict__ Yl,
    const float* __restrict__ nX, const float* __restrict__ nY,
    float* __restrict__ accum) {
  const int it = blockIdx.x, jt = blockIdx.y, z = blockIdx.z;
  if (jt < it) return;
  const uint4* __restrict__ H   = z ? Yh : Xh;
  const uint4* __restrict__ L   = z ? Yl : Xl;
  const float* __restrict__ nrm = z ? nY : nX;

  const int tid   = threadIdx.x;
  const int wave  = tid >> 5;
  const int lane  = tid & 31;
  const int lhalf = lane >> 4;   // 0 or 1
  const int llow  = lane & 15;
  const int i0 = it * 128, j0 = jt * 128;
  const int mrow = (wave & 3) * 32;   // local row block this wave computes
  const int ncol = (wave >> 2) * 64;  // local col block this wave computes

  // Tiles in LDS: [Ah | Al | Bh | Bl], each 128 rows x 4 uint4 = 512 uint4.
  __shared__ uint4 smem[4 * 512];
  __shared__ float red[8];

  // Staging map: thread covers q = tid + 256*i (i<8); tile=q>>9, row=(q>>2)&127,
  // kpart=q&3. Global: plane[(rowbase+row)*64 + ks*4 + kpart] (row = 64 uint4).
  const uint4* gp[8];
  size_t gq[8];
#pragma unroll
  for (int i = 0; i < 8; ++i) {
    const int q    = tid + 256 * i;
    const int tile = q >> 9;
    const int r    = (q >> 2) & 127;
    const int kp   = q & 3;
    const int rowbase = (tile < 2) ? i0 : j0;
    gp[i] = (tile & 1) ? L : H;
    gq[i] = (size_t)(rowbase + r) * 64 + kp;
  }

  v8f acc[2][4] = {};

  uint4 stage[8];
#pragma unroll
  for (int i = 0; i < 8; ++i) stage[i] = gp[i][gq[i]];   // prefetch ks = 0

  for (int ks = 0; ks < KSTEPS; ++ks) {
    if (ks) __syncthreads();            // prior compute done before overwrite
#pragma unroll
    for (int i = 0; i < 8; ++i) smem[tid + 256 * i] = stage[i];
    if (ks + 1 < KSTEPS) {              // issue next k-step; overlaps WMMAs
#pragma unroll
      for (int i = 0; i < 8; ++i) stage[i] = gp[i][gq[i] + (size_t)(ks + 1) * 4];
    }
    __syncthreads();                    // staged tiles visible

    // A operand: lane holds row M; K groups [kb..kb+7],[kb+16..kb+23], kb=8*lhalf.
    v16bf aHi[2], aLo[2];
#pragma unroll
    for (int m = 0; m < 2; ++m) {
      const int ra = (mrow + 16 * m + llow) * 4 + lhalf;
      aHi[m] = frag_from(smem[ra],       smem[ra + 2]);
      aLo[m] = frag_from(smem[512 + ra], smem[512 + ra + 2]);
    }
    // B operand: lane holds column N (= row of x); K groups [kb..kb+7],[kb+16..kb+23], kb=8*lhalf.
#pragma unroll
    for (int n = 0; n < 4; ++n) {
      const int rb = (ncol + 16 * n + llow) * 4 + lhalf;
      const v16bf bHi = frag_from(smem[1024 + rb], smem[1024 + rb + 2]);
      const v16bf bLo = frag_from(smem[1536 + rb], smem[1536 + rb + 2]);
#pragma unroll
      for (int m = 0; m < 2; ++m) {
        acc[m][n] = __builtin_amdgcn_wmma_f32_16x16x32_bf16(
            false, aHi[m], false, bHi, (short)0, acc[m][n], false, false);
        acc[m][n] = __builtin_amdgcn_wmma_f32_16x16x32_bf16(
            false, aHi[m], false, bLo, (short)0, acc[m][n], false, false);
        acc[m][n] = __builtin_amdgcn_wmma_f32_16x16x32_bf16(
            false, aLo[m], false, bHi, (short)0, acc[m][n], false, false);
      }
    }
  }

  // C layout: VGPR r, lanes 0-15 -> M=r, lanes 16-31 -> M=r+8; N = lane&15.
  float ni[2][8];
  float njv[4];
#pragma unroll
  for (int m = 0; m < 2; ++m)
#pragma unroll
    for (int r = 0; r < 8; ++r)
      ni[m][r] = nrm[i0 + mrow + 16 * m + 8 * lhalf + r];
#pragma unroll
  for (int n = 0; n < 4; ++n)
    njv[n] = nrm[j0 + ncol + 16 * n + llow];

  float local = 0.0f;
#pragma unroll
  for (int m = 0; m < 2; ++m)
#pragma unroll
    for (int n = 0; n < 4; ++n)
#pragma unroll
      for (int r = 0; r < 8; ++r) {
        float d2 = ni[m][r] + njv[n] - 2.0f * acc[m][n][r];
        d2 = fmaxf(d2, 0.0f);
        local += __expf(-2.0f * d2);   // T = 2
      }
  if (it != jt) local *= 2.0f;  // symmetric off-diagonal tile counted twice

#pragma unroll
  for (int off = 16; off > 0; off >>= 1) local += __shfl_down(local, off, 32);
  if (lane == 0) red[wave] = local;
  __syncthreads();
  if (tid == 0) {
    float s = 0.0f;
#pragma unroll
    for (int wv = 0; wv < 8; ++wv) s += red[wv];
    atomicAdd(accum + z, s);
  }
}

__global__ __launch_bounds__(32) void finalize_kernel(const float* __restrict__ accum,
                                                      float* __restrict__ out) {
  if (threadIdx.x != 0) return;
  const float n = (float)N_ROWS;
  const float denom = n * (n - 1.0f);        // 2 * npairs
  float sx = accum[0], sy = accum[1], al = accum[2];
  float ux = __logf((sx - n) / denom);
  float uy = __logf((sy - n) / denom);
  out[0] = al / n + 0.5f * (ux + uy);
}

extern "C" void kernel_launch(void* const* d_in, const int* in_sizes, int n_in,
                              void* d_out, int out_size, void* d_ws, size_t ws_size,
                              hipStream_t stream) {
  const float* x = (const float*)d_in[0];
  const float* y = (const float*)d_in[1];
  float* out = (float*)d_out;

  // Workspace layout (16B-aligned sections):
  //   accum[4] | normX[8192] | normY[8192] | Xh | Xl | Yh | Yl  (uint4 planes)
  float* w     = (float*)d_ws;
  float* accum = w;
  float* normX = w + 16;
  float* normY = normX + N_ROWS;
  uint4* xh = (uint4*)(normY + N_ROWS);
  uint4* xl = xh + FRAG_CNT;
  uint4* yh = xl + FRAG_CNT;
  uint4* yl = yh + FRAG_CNT;

  init_kernel<<<1, 32, 0, stream>>>(accum);
  split_kernel<<<FRAG_CNT / 256, 256, 0, stream>>>(x, xh, xl);
  split_kernel<<<FRAG_CNT / 256, 256, 0, stream>>>(y, yh, yl);
  norm_align_kernel<<<N_ROWS, 128, 0, stream>>>(x, y, normX, normY, accum);
  dim3 grid(N_ROWS / 128, N_ROWS / 128, 2);
  gram_exp_kernel<<<grid, 256, 0, stream>>>(xh, xl, yh, yl, normX, normY, accum);
  finalize_kernel<<<1, 32, 0, stream>>>(accum, out);
}
